// SchNet_2774548873968
// MI455X (gfx1250) — compile-verified, matrix operand order checked by reading, not run on hardware
//
#include <hip/hip_runtime.h>
#include <hip/hip_bf16.h>
#include <math.h>

// Problem constants (match reference)
#define BB 16
#define AA 256
#define NN 64
#define FF 128
#define GG 25
#define LL 3
#define CUTOFF 5.0f
#define START 1.2f

typedef __attribute__((ext_vector_type(16))) _Float16 v16h;
typedef __attribute__((ext_vector_type(8)))  float    v8f;

// ---------- helpers ----------

// shifted softplus: log(1+e^x) - log(2), numerically stable
__device__ __forceinline__ float ssp(float v) {
    return fmaxf(v, 0.0f) + log1pf(expf(-fabsf(v))) - 0.69314718056f;
}

// Load a 16-half fragment: elements 0..7 at rowp[0..7], 8..15 at rowp[16..23].
// Caller passes rowp already offset by (row_stride*row + k0 + half*8).
// Matches the CDNA5 16-bit A layout: lanes0-15 K=0..7,16..23; lanes16-31 +8.
__device__ __forceinline__ v16h load_frag_f16(const _Float16* rowp) {
    union { v16h v; uint4 q[2]; } u;
    u.q[0] = *reinterpret_cast<const uint4*>(rowp);
    u.q[1] = *reinterpret_cast<const uint4*>(rowp + 16);
    return u.v;
}

// Build a fragment from an f32 row (convert to f16 on the fly).
__device__ __forceinline__ v16h cvt_frag_f32(const float* row, int k0, int half) {
    v16h r;
#pragma unroll
    for (int e = 0; e < 16; ++e) {
        int k = k0 + (e & 7) + ((e >> 3) << 4) + half * 8;
        r[e] = (_Float16)row[k];
    }
    return r;
}

__device__ __forceinline__ v8f bcast8(float v) {
    v8f c = { v, v, v, v, v, v, v, v };
    return c;
}

// CDNA5 async global->LDS copy of one 16B chunk (per lane), tracked on ASYNCcnt.
__device__ __forceinline__ void async_copy_b128(const _Float16* gsrc, _Float16* lds_dst) {
    uint32_t lo = (uint32_t)(uintptr_t)lds_dst;   // low 32 bits = LDS offset
    asm volatile("global_load_async_to_lds_b128 %0, %1, off"
                 :: "v"(lo), "v"(gsrc) : "memory");
}

__device__ __forceinline__ void wait_asynccnt0() {
    asm volatile("s_wait_asynccnt 0" ::: "memory");
}

// ---------- prep kernels ----------

// src [L][K][N] f32 -> dst [L][N][Kpad] f16 (zero-padded in K)
__global__ void transpose_w_kernel(const float* __restrict__ src, _Float16* __restrict__ dst,
                                   int Lt, int K, int Nn, int Kpad) {
    int idx = blockIdx.x * 256 + threadIdx.x;
    int total = Lt * Nn * Kpad;
    if (idx >= total) return;
    int kp = idx % Kpad;
    int rest = idx / Kpad;
    int n = rest % Nn;
    int l = rest / Nn;
    float v = (kp < K) ? src[((size_t)l * K + kp) * Nn + n] : 0.0f;
    dst[idx] = (_Float16)v;
}

// per (b,i,n): distance -> gaussian smearing (f16, padded to 32) + fcut
__global__ void prep_pairs_kernel(const float* __restrict__ pos, const int* __restrict__ nbr,
                                  const float* __restrict__ mask,
                                  _Float16* __restrict__ fijh, float* __restrict__ fcut) {
    int idx = blockIdx.x * 256 + threadIdx.x;
    if (idx >= BB * AA * NN) return;
    int ba = idx / NN;                 // b*A + i
    int b  = ba / AA;
    int j  = nbr[idx];
    const float* pi = pos + (size_t)ba * 3;
    const float* pj = pos + ((size_t)b * AA + j) * 3;
    float dx = pj[0] - pi[0];
    float dy = pj[1] - pi[1];
    float dz = pj[2] - pi[2];
    float r = sqrtf(dx * dx + dy * dy + dz * dz + 1e-9f);
    const float width = (CUTOFF - START) / (GG - 1);
    const float coef  = -0.5f / (width * width);
    _Float16* out = fijh + (size_t)idx * 32;
#pragma unroll
    for (int g = 0; g < 32; ++g) {
        float v = 0.0f;
        if (g < GG) {
            float d = r - (START + g * width);
            v = expf(coef * d * d);
        }
        out[g] = (_Float16)v;
    }
    fcut[idx] = (r <= CUTOFF ? 1.0f : 0.0f) * mask[idx];
}

// x = embedding[Z] (f32 into d_out) and f16 mirror xh
__global__ void init_x_kernel(const float* __restrict__ emb, const int* __restrict__ Z,
                              float* __restrict__ x, _Float16* __restrict__ xh) {
    int idx = blockIdx.x * 256 + threadIdx.x;
    if (idx >= BB * AA * FF) return;
    int ba = idx / FF;
    int f  = idx % FF;
    float v = emb[(size_t)Z[ba] * FF + f];
    x[idx]  = v;
    xh[idx] = (_Float16)v;
}

// ---------- y = x @ in2f_w[l]  (4096x128 @ 128x128, WMMA) ----------
// grid = B*A/16 blocks of 256 threads; wave w computes tile (mt=blockIdx, nt=w)
__global__ __launch_bounds__(256) void gemm_y_kernel(const _Float16* __restrict__ xh,
                                                     const _Float16* __restrict__ wT,
                                                     float* __restrict__ y) {
    int lane = threadIdx.x & 31, wave = threadIdx.x >> 5;
    int half = lane >> 4, m = lane & 15;
    int mt = blockIdx.x;
    int nt = wave;
    int col = nt * 16 + m;
    const _Float16* xrow = xh + (size_t)(mt * 16 + m) * FF;
    v8f c = bcast8(0.0f);
#pragma unroll
    for (int kc = 0; kc < 4; ++kc) {
        v16h af = load_frag_f16(xrow + kc * 32 + half * 8);
        v16h bf = load_frag_f16(wT + (size_t)col * FF + kc * 32 + half * 8);
        c = __builtin_amdgcn_wmma_f32_16x16x32_f16(false, af, false, bf, (short)0, c, false, false);
    }
#pragma unroll
    for (int r = 0; r < 8; ++r)
        y[((size_t)mt * 16 + r + half * 8) * FF + col] = c[r];
}

// ---------- fused interaction layer ----------
// one block = (molecule b, 16-atom tile). 256 threads = 8 waves, 2 atoms/wave.
__global__ __launch_bounds__(256) void fused_interaction_kernel(
        const _Float16* __restrict__ fijh, const float* __restrict__ fcut,
        const float* __restrict__ y, float* __restrict__ x, _Float16* __restrict__ xh,
        const int* __restrict__ neighbors,
        const _Float16* __restrict__ w1T, const _Float16* __restrict__ w2T,
        const _Float16* __restrict__ f2outT, const _Float16* __restrict__ denseT,
        const float* __restrict__ b1, const float* __restrict__ b2,
        const float* __restrict__ f2b, const float* __restrict__ db) {

    // LDS-resident filter-network weights (hot: reused by every wave 8x)
    __shared__ __align__(16) _Float16 w1Lds[FF * 32];     //  8 KB
    __shared__ __align__(16) _Float16 w2Lds[FF * FF];     // 32 KB
    __shared__ __align__(16) _Float16 HLds[8][16][136];   // ~34 KB per-wave H row-block
    __shared__ int   nbrLds[16 * NN];                     //  4 KB
    __shared__ float fcutLds[16 * NN];                    //  4 KB
    __shared__ float aggLds[16][FF];                      //  8 KB
    __shared__ __align__(16) _Float16 t1Lds[16][136];     // ~4 KB

    const int lane = threadIdx.x & 31, wave = threadIdx.x >> 5;
    const int half = lane >> 4, m = lane & 15;
    const int bmol = blockIdx.x >> 4;            // A/16 = 16 tiles per molecule
    const int atomBase = (blockIdx.x & 15) * 16;
    const size_t baseRow = (size_t)bmol * AA + atomBase;   // row into x/y ([B*A, F])

    // async-stage w1T/w2T into LDS (CDNA5 async-to-LDS path, ASYNCcnt-tracked)
    for (int c = threadIdx.x; c < (FF * 32) / 8; c += 256)
        async_copy_b128(w1T + c * 8, &w1Lds[c * 8]);
    for (int c = threadIdx.x; c < (FF * FF) / 8; c += 256)
        async_copy_b128(w2T + c * 8, &w2Lds[c * 8]);

    // cooperative load of neighbor indices + cutoff for this atom tile
    for (int p = threadIdx.x; p < 16 * NN; p += 256) {
        int a = p >> 6, n = p & 63;
        size_t gp = (baseRow + a) * NN + n;
        nbrLds[p]  = neighbors[gp];
        fcutLds[p] = fcut[gp];
    }
    wait_asynccnt0();
    __syncthreads();

    float agg[8];
#pragma unroll
    for (int nt = 0; nt < 8; ++nt) agg[nt] = 0.0f;

    // 8 uniform iterations: atom = wave*2 + iter/4, row-block rb = iter%4
    for (int iter = 0; iter < 8; ++iter) {
        const int a  = wave * 2 + (iter >> 2);
        const int rb = iter & 3;
        const int rowLocal = a * NN + rb * 16;
        const size_t pairRow = (baseRow + a) * NN + rb * 16 + m;

        // step 1: H = ssp(f_ij @ w1 + b1) -> f16 into per-wave LDS
        v16h afij = load_frag_f16(fijh + pairRow * 32 + half * 8);
#pragma unroll
        for (int nt = 0; nt < 8; ++nt) {
            int col = nt * 16 + m;
            v8f c = bcast8(b1[col]);
            v16h bf = load_frag_f16(&w1Lds[col * 32 + half * 8]);
            c = __builtin_amdgcn_wmma_f32_16x16x32_f16(false, afij, false, bf, (short)0, c, false, false);
#pragma unroll
            for (int r = 0; r < 8; ++r)
                HLds[wave][r + half * 8][col] = (_Float16)ssp(c[r]);
        }
        __syncthreads();

        // reload H as A fragments (4 K-chunks)
        v16h ha[4];
#pragma unroll
        for (int kc = 0; kc < 4; ++kc)
            ha[kc] = load_frag_f16(&HLds[wave][m][0] + kc * 32 + half * 8);

        // step 2: W = H @ w2 + b2, mask, gather y[nbr], accumulate agg
#pragma unroll
        for (int nt = 0; nt < 8; ++nt) {
            int col = nt * 16 + m;
            v8f c = bcast8(b2[col]);
#pragma unroll
            for (int kc = 0; kc < 4; ++kc) {
                v16h bf = load_frag_f16(&w2Lds[col * FF + kc * 32 + half * 8]);
                c = __builtin_amdgcn_wmma_f32_16x16x32_f16(false, ha[kc], false, bf, (short)0, c, false, false);
            }
#pragma unroll
            for (int r = 0; r < 8; ++r) {
                int rl = rowLocal + r + half * 8;
                float wv = c[r] * fcutLds[rl];
                int nb = nbrLds[rl];
                float yv = y[((size_t)bmol * AA + nb) * FF + col];
                agg[nt] = fmaf(wv, yv, agg[nt]);
            }
        }

        // after the atom's last row-block: reduce halves, stash agg
        if (rb == 3) {
#pragma unroll
            for (int nt = 0; nt < 8; ++nt) {
                float v = agg[nt];
                v += __shfl_xor(v, 16, 32);
                if (half == 0) aggLds[a][nt * 16 + m] = v;
                agg[nt] = 0.0f;
            }
        }
    }
    __syncthreads();

    // step 4a: t1 = ssp(agg @ f2out + f2out_b)   (16 atoms x 128), nt = wave
    {
        int nt = wave;
        int col = nt * 16 + m;
        v8f c = bcast8(f2b[col]);
#pragma unroll
        for (int kc = 0; kc < 4; ++kc) {
            v16h af = cvt_frag_f32(&aggLds[m][0], kc * 32, half);
            v16h bf = load_frag_f16(f2outT + (size_t)col * FF + kc * 32 + half * 8);
            c = __builtin_amdgcn_wmma_f32_16x16x32_f16(false, af, false, bf, (short)0, c, false, false);
        }
#pragma unroll
        for (int r = 0; r < 8; ++r)
            t1Lds[r + half * 8][col] = (_Float16)ssp(c[r]);
    }
    __syncthreads();

    // step 4b: v = t1 @ dense + dense_b ; x += v (and refresh f16 mirror)
    {
        int nt = wave;
        int col = nt * 16 + m;
        v8f c = bcast8(db[col]);
#pragma unroll
        for (int kc = 0; kc < 4; ++kc) {
            v16h af = load_frag_f16(&t1Lds[m][0] + kc * 32 + half * 8);
            v16h bf = load_frag_f16(denseT + (size_t)col * FF + kc * 32 + half * 8);
            c = __builtin_amdgcn_wmma_f32_16x16x32_f16(false, af, false, bf, (short)0, c, false, false);
        }
#pragma unroll
        for (int r = 0; r < 8; ++r) {
            size_t xo = (baseRow + r + half * 8) * FF + col;
            float nx = x[xo] + c[r];
            x[xo]  = nx;
            xh[xo] = (_Float16)nx;
        }
    }
}

// ---------- host launcher ----------

extern "C" void kernel_launch(void* const* d_in, const int* in_sizes, int n_in,
                              void* d_out, int out_size, void* d_ws, size_t ws_size,
                              hipStream_t stream) {
    const float* embedding = (const float*)d_in[0];
    const float* positions = (const float*)d_in[1];
    const float* filt_w1   = (const float*)d_in[2];
    const float* filt_b1   = (const float*)d_in[3];
    const float* filt_w2   = (const float*)d_in[4];
    const float* filt_b2   = (const float*)d_in[5];
    const float* in2f_w    = (const float*)d_in[6];
    const float* f2out_w   = (const float*)d_in[7];
    const float* f2out_b   = (const float*)d_in[8];
    const float* dense_w   = (const float*)d_in[9];
    const float* dense_b   = (const float*)d_in[10];
    const int*   Z         = (const int*)d_in[11];
    const int*   neighbors = (const int*)d_in[12];
    const float* nbr_mask  = (const float*)d_in[13];
    float* x = (float*)d_out;

    // workspace carve-up (256B aligned chunks)
    char* ws = (char*)d_ws;
    auto carve = [&](size_t bytes) {
        char* p = ws;
        ws += (bytes + 255) & ~(size_t)255;
        return p;
    };
    _Float16* fijh   = (_Float16*)carve((size_t)BB * AA * NN * 32 * sizeof(_Float16));
    float*    fcut   = (float*)   carve((size_t)BB * AA * NN * sizeof(float));
    float*    y      = (float*)   carve((size_t)BB * AA * FF * sizeof(float));
    _Float16* xh     = (_Float16*)carve((size_t)BB * AA * FF * sizeof(_Float16));
    _Float16* w1T    = (_Float16*)carve((size_t)LL * FF * 32 * sizeof(_Float16));
    _Float16* w2T    = (_Float16*)carve((size_t)LL * FF * FF * sizeof(_Float16));
    _Float16* in2fT  = (_Float16*)carve((size_t)LL * FF * FF * sizeof(_Float16));
    _Float16* f2outT = (_Float16*)carve((size_t)LL * FF * FF * sizeof(_Float16));
    _Float16* denseT = (_Float16*)carve((size_t)LL * FF * FF * sizeof(_Float16));

    // weight transposition / f16 conversion
    {
        int t1 = LL * FF * 32;
        transpose_w_kernel<<<(t1 + 255) / 256, 256, 0, stream>>>(filt_w1, w1T, LL, GG, FF, 32);
        int t2 = LL * FF * FF;
        transpose_w_kernel<<<(t2 + 255) / 256, 256, 0, stream>>>(filt_w2, w2T, LL, FF, FF, FF);
        transpose_w_kernel<<<(t2 + 255) / 256, 256, 0, stream>>>(in2f_w, in2fT, LL, FF, FF, FF);
        transpose_w_kernel<<<(t2 + 255) / 256, 256, 0, stream>>>(f2out_w, f2outT, LL, FF, FF, FF);
        transpose_w_kernel<<<(t2 + 255) / 256, 256, 0, stream>>>(dense_w, denseT, LL, FF, FF, FF);
    }
    // RBF features + cutoff
    prep_pairs_kernel<<<(BB * AA * NN + 255) / 256, 256, 0, stream>>>(positions, neighbors,
                                                                      nbr_mask, fijh, fcut);
    // x = embedding[Z] directly into d_out (+ f16 mirror)
    init_x_kernel<<<(BB * AA * FF + 255) / 256, 256, 0, stream>>>(embedding, Z, x, xh);

    for (int l = 0; l < LL; ++l) {
        gemm_y_kernel<<<BB * AA / 16, 256, 0, stream>>>(xh, in2fT + (size_t)l * FF * FF, y);
        fused_interaction_kernel<<<BB * AA / 16, 256, 0, stream>>>(
            fijh, fcut, y, x, xh, neighbors,
            w1T + (size_t)l * FF * 32, w2T + (size_t)l * FF * FF,
            f2outT + (size_t)l * FF * FF, denseT + (size_t)l * FF * FF,
            filt_b1 + (size_t)l * FF, filt_b2 + (size_t)l * FF,
            f2out_b + (size_t)l * FF, dense_b + (size_t)l * FF);
    }
}
